// DecoderWithAttention_83245056131855
// MI455X (gfx1250) — compile-verified, matrix operand order checked by reading, not run on hardware
//
#include <hip/hip_runtime.h>
#include <math.h>

// ---------------- problem sizes ----------------
#define Bsz 128
#define Psz 196
#define Esz 2048
#define Dsz 512
#define Asz 512
#define EMsz 512
#define Vsz 10000
#define Vpad 10048         // 157 * 64, padded vocab for branch-free GEMM
#define Lsz 52
#define Tsz (Lsz - 1)      // 51
#define XK (EMsz + Esz)    // 2560
#define G4 (4 * Dsz)       // 2048

typedef __attribute__((ext_vector_type(16))) __bf16 v16bf;
typedef __attribute__((ext_vector_type(8)))  float  v8f;

__device__ __forceinline__ float sigf(float x) { return 1.f / (1.f + __expf(-x)); }

// Load one 16x32 bf16 fragment (A or B operand of v_wmma_f32_16x16x32_bf16).
// Per ISA 7.12.2: lane<16 holds row=lane, K={kh*8+0..7, 16+kh*8+0..7}, kh = lane>>4.
// Caller passes p already offset by (row*ld + kh*8 + kblock).
__device__ __forceinline__ v16bf load_frag(const __bf16* p) {
  v16bf f;
#pragma unroll
  for (int i = 0; i < 8; ++i) f[i] = p[i];
#pragma unroll
  for (int i = 0; i < 8; ++i) f[8 + i] = p[16 + i];
  return f;
}

// ---------------- generic bf16 WMMA GEMM (branch-free mainloop) ----------------
// C[m,n] = act( sum_k A[m,k]*B[n,k] + bias[n] (+ C_old if accum) )
// Requires: M % 128 == 0, N % 64 == 0, K % 32 == 0.
// Block = 128 threads = 4 waves; each wave computes a 32x64 tile
// (2 M-subtiles x 4 N-subtiles = 8 WMMAs per K step, B fragments reused).
__global__ void k_gemm(const __bf16* __restrict__ Am, int lda,
                       const __bf16* __restrict__ Bm, int ldb,
                       const float* __restrict__ bias,
                       float* Cf, __bf16* Cb, long ldc,
                       int K, int accum, int act) {
  const int wave = threadIdx.x >> 5;
  const int lane = threadIdx.x & 31;
  const int m0 = blockIdx.y * 128 + wave * 32;
  const int n0 = blockIdx.x * 64;
  const int row = lane & 15;
  const int kh  = lane >> 4;

  v8f zero = {0.f, 0.f, 0.f, 0.f, 0.f, 0.f, 0.f, 0.f};
  v8f acc[2][4];
#pragma unroll
  for (int s = 0; s < 2; ++s)
#pragma unroll
    for (int i = 0; i < 4; ++i) acc[s][i] = zero;

  const __bf16* ap0 = Am + (long)(m0 + row) * lda + kh * 8;
  const __bf16* ap1 = ap0 + (long)16 * lda;
  const __bf16* bp  = Bm + (long)(n0 + row) * ldb + kh * 8;

  for (int kk = 0; kk < K; kk += 32) {
    v16bf a0 = load_frag(ap0 + kk);
    v16bf a1 = load_frag(ap1 + kk);
    __builtin_prefetch(ap0 + kk + 128, 0, 0);   // global_prefetch_b8
#pragma unroll
    for (int i = 0; i < 4; ++i) {
      v16bf bf = load_frag(bp + (long)i * 16 * ldb + kk);
      acc[0][i] = __builtin_amdgcn_wmma_f32_16x16x32_bf16(
          false, a0, false, bf, (short)0, acc[0][i], false, false);
      acc[1][i] = __builtin_amdgcn_wmma_f32_16x16x32_bf16(
          false, a1, false, bf, (short)0, acc[1][i], false, false);
    }
  }

  // C fragment layout: lane<16 -> N=lane, VGPR r -> M=r; lane>=16 -> N=lane-16, M=8+r
  const int n_lane = lane & 15;
  const int m_off  = (lane >> 4) * 8;
#pragma unroll
  for (int s = 0; s < 2; ++s) {
#pragma unroll
    for (int i = 0; i < 4; ++i) {
      int n = n0 + i * 16 + n_lane;
      float bv = bias ? bias[n] : 0.f;
#pragma unroll
      for (int r = 0; r < 8; ++r) {
        long m = m0 + s * 16 + m_off + r;
        float v = acc[s][i][r] + bv;
        if (accum) v += Cf[m * ldc + n];
        if (act == 1) v = 1.f / (1.f + __expf(-v));
        if (Cb) Cb[m * ldc + n] = (__bf16)v;
        else    Cf[m * ldc + n] = v;
      }
    }
  }
}

// ---------------- small helper kernels ----------------
__global__ void k_cvt(const float* __restrict__ s, __bf16* __restrict__ d, long n) {
  for (long i = (long)blockIdx.x * blockDim.x + threadIdx.x; i < n;
       i += (long)gridDim.x * blockDim.x)
    d[i] = (__bf16)s[i];
}

__global__ void k_zero_bf16(__bf16* __restrict__ d, long n) {
  for (long i = (long)blockIdx.x * blockDim.x + threadIdx.x; i < n;
       i += (long)gridDim.x * blockDim.x)
    d[i] = (__bf16)0.f;
}

// argsort(-len) stable + outputs: encoded_captions (float copy), decode_lengths, sort_ind
__global__ void k_sort(const long long* __restrict__ cl, const long long* __restrict__ enc,
                       float* ec_out, float* dl_out, float* si_out, int* dlen_ws) {
  __shared__ long long s_cl[Bsz];
  int i = threadIdx.x;
  if (i < Bsz) s_cl[i] = cl[i];
  __syncthreads();
  if (i < Bsz) {
    long long v = s_cl[i];
    int rank = 0;
    for (int j = 0; j < Bsz; ++j) {
      long long u = s_cl[j];
      if (u > v || (u == v && j < i)) ++rank;
    }
    si_out[rank]  = (float)i;
    dl_out[rank]  = (float)(v - 1);
    dlen_ws[rank] = (int)(v - 1);
  }
  for (int j = i; j < Bsz * Lsz; j += blockDim.x) ec_out[j] = (float)enc[j];
}

__global__ void k_mask(const int* __restrict__ dlen, int t, int* mask) {
  int b = threadIdx.x;
  if (b < Bsz) mask[b] = (dlen[b] > t) ? 1 : 0;
}

__global__ void k_mean(const float* __restrict__ out, float* __restrict__ mo) {
  for (long idx = (long)blockIdx.x * blockDim.x + threadIdx.x; idx < (long)Bsz * Esz;
       idx += (long)gridDim.x * blockDim.x) {
    long b = idx / Esz, e = idx % Esz;
    const float* p = out + b * (long)Psz * Esz + e;
    float s = 0.f;
    for (int pp = 0; pp < Psz; ++pp) s += p[(long)pp * Esz];
    mo[idx] = s * (1.f / Psz);
  }
}

__global__ void k_h0c0(const float* __restrict__ mo,
                       const float* __restrict__ Wh0, const float* __restrict__ bh0,
                       const float* __restrict__ Wc0, const float* __restrict__ bc0,
                       float* h, float* c, __bf16* hb) {
  for (int idx = blockIdx.x * blockDim.x + threadIdx.x; idx < Bsz * Dsz;
       idx += gridDim.x * blockDim.x) {
    int b = idx / Dsz, d = idx % Dsz;
    const float* m  = mo + (long)b * Esz;
    const float* wh = Wh0 + (long)d * Esz;
    const float* wc = Wc0 + (long)d * Esz;
    float sh = 0.f, sc = 0.f;
    for (int e = 0; e < Esz; ++e) { float mv = m[e]; sh += mv * wh[e]; sc += mv * wc[e]; }
    float hv = sh + bh0[d];
    h[idx] = hv; c[idx] = sc + bc0[d]; hb[idx] = (__bf16)hv;
  }
}

// scores = relu(att1 + att2) . wa + ba ; softmax over P ; masked alphas store
__global__ void k_scores(const __bf16* __restrict__ att1b, const float* __restrict__ att2,
                         const float* __restrict__ wa, const float* __restrict__ ba,
                         float* __restrict__ alpha_ws, float* __restrict__ alphas_out,
                         const int* __restrict__ mask, int t) {
  const int b = blockIdx.x, tid = threadIdx.x;
  __shared__ float s_a2[Asz];
  __shared__ float s_wa[Asz];
  __shared__ float s_sc[Psz];
  __shared__ float s_red[256];
  for (int a = tid; a < Asz; a += 256) { s_a2[a] = att2[b * Asz + a]; s_wa[a] = wa[a]; }
  __syncthreads();
  for (int p = tid; p < Psz; p += 256) {
    const __bf16* rowp = att1b + ((long)b * Psz + p) * Asz;
    float s = 0.f;
    for (int a = 0; a < Asz; ++a) {
      float v = (float)rowp[a] + s_a2[a];
      v = v > 0.f ? v : 0.f;
      s += v * s_wa[a];
    }
    s_sc[p] = s + ba[0];
  }
  __syncthreads();
  float mx = -3.4e38f;
  for (int p = tid; p < Psz; p += 256) mx = fmaxf(mx, s_sc[p]);
  s_red[tid] = mx; __syncthreads();
  for (int o = 128; o > 0; o >>= 1) { if (tid < o) s_red[tid] = fmaxf(s_red[tid], s_red[tid + o]); __syncthreads(); }
  mx = s_red[0]; __syncthreads();
  float sm = 0.f;
  for (int p = tid; p < Psz; p += 256) sm += __expf(s_sc[p] - mx);
  s_red[tid] = sm; __syncthreads();
  for (int o = 128; o > 0; o >>= 1) { if (tid < o) s_red[tid] += s_red[tid + o]; __syncthreads(); }
  float inv = 1.f / s_red[0];
  int mk = mask[b];
  for (int p = tid; p < Psz; p += 256) {
    float al = __expf(s_sc[p] - mx) * inv;
    alpha_ws[b * Psz + p] = al;
    alphas_out[((long)b * Tsz + t) * Psz + p] = mk ? al : 0.f;
  }
}

// awe = sum_p alpha*out ; x = [emb_t, gate*awe] (bf16)
__global__ void k_awe_x(const __bf16* __restrict__ outb, const float* __restrict__ alpha_ws,
                        const float* __restrict__ gate, const float* __restrict__ Wemb,
                        const long long* __restrict__ enc, int t, __bf16* __restrict__ xb) {
  const int b = blockIdx.x, tid = threadIdx.x;
  __shared__ float s_al[Psz];
  for (int p = tid; p < Psz; p += 256) s_al[p] = alpha_ws[b * Psz + p];
  __syncthreads();
  float acc[8];
#pragma unroll
  for (int j = 0; j < 8; ++j) acc[j] = 0.f;
  const int e0 = tid * 8;                        // 256 threads * 8 = 2048 = E
  for (int p = 0; p < Psz; ++p) {
    float al = s_al[p];
    const __bf16* rp = outb + ((long)b * Psz + p) * Esz + e0;
#pragma unroll
    for (int j = 0; j < 8; ++j) acc[j] += al * (float)rp[j];
  }
  const float* g = gate + (long)b * Esz + e0;
  __bf16* xr = xb + (long)b * XK + EMsz + e0;
#pragma unroll
  for (int j = 0; j < 8; ++j) xr[j] = (__bf16)(acc[j] * g[j]);
  long tok = (long)enc[(long)b * Lsz + t];
  const float* emb = Wemb + tok * (long)EMsz;
  __bf16* xe = xb + (long)b * XK;
  for (int j = tid; j < EMsz; j += 256) xe[j] = (__bf16)emb[j];
}

__global__ void k_lstm(const float* __restrict__ gates, float* h, float* c,
                       __bf16* hb, __bf16* hnb, const int* __restrict__ mask) {
  for (int idx = blockIdx.x * blockDim.x + threadIdx.x; idx < Bsz * Dsz;
       idx += gridDim.x * blockDim.x) {
    int b = idx / Dsz, d = idx % Dsz;
    const float* gr = gates + (long)b * G4;
    float gi = gr[d], gf = gr[Dsz + d], gg = gr[2 * Dsz + d], go = gr[3 * Dsz + d];
    float cn = sigf(gf) * c[idx] + sigf(gi) * tanhf(gg);
    float hn = sigf(go) * tanhf(cn);
    hnb[idx] = (__bf16)hn;
    if (mask[b]) { h[idx] = hn; c[idx] = cn; }
    hb[idx] = (__bf16)h[idx];
  }
}

// preds epilogue: add bias, apply row mask, store 10000 valid cols of padded tmp
__global__ void k_pred_store(const float* __restrict__ ptmp, const float* __restrict__ bfc,
                             const int* __restrict__ mask, int t, float* __restrict__ outp) {
  for (long idx = (long)blockIdx.x * blockDim.x + threadIdx.x; idx < (long)Bsz * Vsz;
       idx += (long)gridDim.x * blockDim.x) {
    long b = idx / Vsz, n = idx % Vsz;
    float v = ptmp[b * Vpad + n] + bfc[n];
    outp[(b * Tsz + t) * (long)Vsz + n] = mask[b] ? v : 0.f;
  }
}

// ---------------- host launcher ----------------
extern "C" void kernel_launch(void* const* d_in, const int* in_sizes, int n_in,
                              void* d_out, int out_size, void* d_ws, size_t ws_size,
                              hipStream_t stream) {
  (void)in_sizes; (void)n_in; (void)out_size; (void)ws_size;
  const float*     out_f = (const float*)d_in[0];
  const long long* enc   = (const long long*)d_in[1];
  const long long* cl    = (const long long*)d_in[2];
  const float* We  = (const float*)d_in[3];
  const float* be  = (const float*)d_in[4];
  const float* Wd  = (const float*)d_in[5];
  const float* bd  = (const float*)d_in[6];
  const float* wa  = (const float*)d_in[7];
  const float* ba  = (const float*)d_in[8];
  const float* Wemb= (const float*)d_in[9];
  const float* Wih = (const float*)d_in[10];
  const float* bih = (const float*)d_in[11];
  const float* Whh = (const float*)d_in[12];
  const float* bhh = (const float*)d_in[13];
  const float* Wh0 = (const float*)d_in[14];
  const float* bh0 = (const float*)d_in[15];
  const float* Wc0 = (const float*)d_in[16];
  const float* bc0 = (const float*)d_in[17];
  const float* Wfb = (const float*)d_in[18];
  const float* bfb = (const float*)d_in[19];
  const float* Wfc = (const float*)d_in[20];
  const float* bfc = (const float*)d_in[21];

  float* outp = (float*)d_out;
  const long pred_off = 0;
  const long ec_off = (long)Bsz * Tsz * Vsz;
  const long dl_off = ec_off + (long)Bsz * Lsz;
  const long al_off = dl_off + Bsz;
  const long si_off = al_off + (long)Bsz * Tsz * Psz;

  char* ws = (char*)d_ws;
  size_t off = 0;
  auto alloc = [&](size_t bytes) -> char* {
    char* p = ws + off; off += (bytes + 255) & ~(size_t)255; return p;
  };
  __bf16* outb  = (__bf16*)alloc((size_t)Bsz * Psz * Esz * 2);  // L2-resident stream
  __bf16* att1b = (__bf16*)alloc((size_t)Bsz * Psz * Asz * 2);  // L2-resident stream
  __bf16* Web   = (__bf16*)alloc((size_t)Asz * Esz * 2);
  __bf16* Wdb   = (__bf16*)alloc((size_t)Asz * Dsz * 2);
  __bf16* Wfbb  = (__bf16*)alloc((size_t)Esz * Dsz * 2);
  __bf16* Wihb  = (__bf16*)alloc((size_t)G4 * XK * 2);
  __bf16* Whhb  = (__bf16*)alloc((size_t)G4 * Dsz * 2);
  __bf16* Wfcb  = (__bf16*)alloc((size_t)Vpad * Dsz * 2);       // padded rows
  float*  ptmp  = (float*)alloc((size_t)Bsz * Vpad * 4);        // padded preds
  float*  mo    = (float*)alloc((size_t)Bsz * Esz * 4);
  float*  h     = (float*)alloc((size_t)Bsz * Dsz * 4);
  float*  c     = (float*)alloc((size_t)Bsz * Dsz * 4);
  __bf16* hb    = (__bf16*)alloc((size_t)Bsz * Dsz * 2);
  __bf16* hnb   = (__bf16*)alloc((size_t)Bsz * Dsz * 2);
  float*  att2  = (float*)alloc((size_t)Bsz * Asz * 4);
  float*  alpha = (float*)alloc((size_t)Bsz * Psz * 4);
  float*  gate  = (float*)alloc((size_t)Bsz * Esz * 4);
  __bf16* xb    = (__bf16*)alloc((size_t)Bsz * XK * 2);
  float*  gates = (float*)alloc((size_t)Bsz * G4 * 4);
  int*    dlen  = (int*)alloc(Bsz * 4);
  int*    maskp = (int*)alloc(Bsz * 4);

  // ---- one-time setup ----
  k_sort<<<1, 128, 0, stream>>>(cl, enc, outp + ec_off, outp + dl_off, outp + si_off, dlen);
  k_cvt<<<2048, 256, 0, stream>>>(out_f, outb, (long)Bsz * Psz * Esz);
  k_cvt<<<512, 256, 0, stream>>>(We, Web, (long)Asz * Esz);
  k_cvt<<<256, 256, 0, stream>>>(Wd, Wdb, (long)Asz * Dsz);
  k_cvt<<<512, 256, 0, stream>>>(Wfb, Wfbb, (long)Esz * Dsz);
  k_cvt<<<2048, 256, 0, stream>>>(Wih, Wihb, (long)G4 * XK);
  k_cvt<<<512, 256, 0, stream>>>(Whh, Whhb, (long)G4 * Dsz);
  k_cvt<<<2048, 256, 0, stream>>>(Wfc, Wfcb, (long)Vsz * Dsz);
  k_zero_bf16<<<48, 256, 0, stream>>>(Wfcb + (long)Vsz * Dsz, (long)(Vpad - Vsz) * Dsz);
  k_mean<<<1024, 256, 0, stream>>>(out_f, mo);
  k_h0c0<<<256, 256, 0, stream>>>(mo, Wh0, bh0, Wc0, bc0, h, c, hb);

  // att1 = out @ We.T + be  (25088 x 512, K=2048) -> bf16
  {
    dim3 g(Asz / 64, (Bsz * Psz) / 128);   // 8 x 196
    k_gemm<<<g, 128, 0, stream>>>(outb, Esz, Web, Esz, be, nullptr, att1b, Asz,
                                  Esz, 0, 0);
  }

  // ---- sequential decode steps ----
  for (int t = 0; t < Tsz; ++t) {
    k_mask<<<1, 128, 0, stream>>>(dlen, t, maskp);
    {   // att2 = h @ Wd.T + bd  (128 x 512, K=512)
      dim3 g(Asz / 64, Bsz / 128);
      k_gemm<<<g, 128, 0, stream>>>(hb, Dsz, Wdb, Dsz, bd, att2, nullptr, Asz,
                                    Dsz, 0, 0);
    }
    k_scores<<<Bsz, 256, 0, stream>>>(att1b, att2, wa, ba, alpha, outp + al_off, maskp, t);
    {   // gate = sigmoid(h @ Wfb.T + bfb)  (128 x 2048, K=512)
      dim3 g(Esz / 64, Bsz / 128);
      k_gemm<<<g, 128, 0, stream>>>(hb, Dsz, Wfbb, Dsz, bfb, gate, nullptr, Esz,
                                    Dsz, 0, 1);
    }
    k_awe_x<<<Bsz, 256, 0, stream>>>(outb, alpha, gate, Wemb, enc, t, xb);
    {   // gates = x @ Wih.T + bih  (128 x 2048, K=2560)
      dim3 g(G4 / 64, Bsz / 128);
      k_gemm<<<g, 128, 0, stream>>>(xb, XK, Wihb, XK, bih, gates, nullptr, G4,
                                    XK, 0, 0);
    }
    {   // gates += h @ Whh.T + bhh  (128 x 2048, K=512)
      dim3 g(G4 / 64, Bsz / 128);
      k_gemm<<<g, 128, 0, stream>>>(hb, Dsz, Whhb, Dsz, bhh, gates, nullptr, G4,
                                    Dsz, 1, 0);
    }
    k_lstm<<<256, 256, 0, stream>>>(gates, h, c, hb, hnb, maskp);
    {   // preds_tmp = h_new @ Wfc_pad.T  (128 x 10048, K=512)
      dim3 g(Vpad / 64, Bsz / 128);   // 157 x 1
      k_gemm<<<g, 128, 0, stream>>>(hnb, Dsz, Wfcb, Dsz, nullptr, ptmp, nullptr,
                                    Vpad, Dsz, 0, 0);
    }
    k_pred_store<<<2048, 256, 0, stream>>>(ptmp, bfc, maskp, t, outp + pred_off);
  }
}
